// PreEmbeddedGraphEncoderWithAttention_6193342841319
// MI455X (gfx1250) — matrix operand order repaired and checked by reading, not compiled, not run on hardware
//
#include <hip/hip_runtime.h>
#include <hip/hip_bf16.h>
#include <math.h>

// ---------------- problem constants ----------------
#define BG   64
#define NN   128
#define EE   2048
#define DCH  128
#define LL   4
#define TN   8192            // BG*NN total nodes

typedef __bf16 bf16_t;
typedef __attribute__((ext_vector_type(16))) __bf16 v16bf;
typedef __attribute__((ext_vector_type(8)))  __bf16 v8bf;
typedef __attribute__((ext_vector_type(8)))  float  v8f;
typedef __attribute__((ext_vector_type(4)))  float  v4f;
typedef __attribute__((ext_vector_type(4)))  unsigned int v4u;

#if defined(__has_builtin)
#if __has_builtin(__builtin_amdgcn_global_load_async_to_lds_b128)
#define HAS_ASYNC_LDS 1
#endif
#endif

#if defined(HAS_ASYNC_LDS)
// builtin expects: (int4(gcc-vector) AS1*, int4 AS3*, imm offset, imm cpol)
typedef int v4i_g __attribute__((vector_size(16)));
typedef __attribute__((address_space(1))) v4i_g* gas_p;
typedef __attribute__((address_space(3))) v4i_g* las_p;
#if __has_builtin(__builtin_amdgcn_s_wait_asynccnt)
#define ASYNC_WAIT(n) __builtin_amdgcn_s_wait_asynccnt(n)
#else
#define ASYNC_WAIT(n) asm volatile("s_wait_asynccnt %0" ::"n"(n) : "memory")
#endif
#endif

static __device__ inline v8f vzero8() {
  v8f z;
#pragma unroll
  for (int i = 0; i < 8; ++i) z[i] = 0.f;
  return z;
}

static __device__ inline v16bf combine16(v8bf lo, v8bf hp) {
  v16bf t;
#pragma unroll
  for (int i = 0; i < 8; ++i) { t[i] = lo[i]; t[8 + i] = hp[i]; }
  return t;
}

// =====================================================================
// Small prep kernels
// =====================================================================

__global__ __launch_bounds__(256) void wsum_kernel(const float* __restrict__ Wc,
                                                   float* __restrict__ Wsum) {
  int i = blockIdx.x * 256 + threadIdx.x;
  if (i < DCH * DCH) Wsum[i] = Wc[i] + Wc[i + DCH * DCH];
}

__global__ __launch_bounds__(128) void w2_kernel(const float* __restrict__ Wo,
                                                 const float* __restrict__ Wsum,
                                                 float* __restrict__ W2) {
  int k = blockIdx.x, n = threadIdx.x;
  float s = 0.f;
  for (int j = 0; j < DCH; ++j) s += Wo[k * DCH + j] * Wsum[j * DCH + n];
  W2[k * DCH + n] = s;
}

__global__ __launch_bounds__(128) void b2_kernel(const float* __restrict__ bo,
                                                 const float* __restrict__ Wsum,
                                                 const float* __restrict__ bc,
                                                 float* __restrict__ b2) {
  int n = threadIdx.x;
  float s = bc[n];
  for (int k = 0; k < DCH; ++k) s += bo[k] * Wsum[k * DCH + n];
  b2[n] = s;
}

__global__ __launch_bounds__(128) void wtrans_kernel(const float* __restrict__ W,
                                                     bf16_t* __restrict__ WT) {
  int k = blockIdx.x, n = threadIdx.x;
  WT[n * DCH + k] = (bf16_t)W[k * DCH + n];
}

__global__ __launch_bounds__(256) void deg_init_kernel(float* __restrict__ deg) {
  int i = blockIdx.x * 256 + threadIdx.x;
  if (i < TN) deg[i] = 1.0f;
}

__global__ __launch_bounds__(256) void deg_scatter_kernel(const int* __restrict__ edges,
                                                          float* __restrict__ deg) {
  int i = blockIdx.x * 256 + threadIdx.x;
  if (i >= BG * EE) return;
  int b = i / EE, e = i - b * EE;
  int dst = edges[b * 2 * EE + EE + e] + b * NN;
  atomicAdd(&deg[dst], 1.0f);
}

__global__ __launch_bounds__(256) void rsqrt_kernel(float* __restrict__ norm) {
  int i = blockIdx.x * 256 + threadIdx.x;
  if (i < TN) norm[i] = rsqrtf(norm[i]);
}

__global__ __launch_bounds__(128) void vtrans_kernel(const float* __restrict__ V,
                                                     bf16_t* __restrict__ VT) {
  int t = blockIdx.x, n = threadIdx.x;
  VT[(size_t)n * TN + t] = (bf16_t)V[(size_t)t * DCH + n];
}

__global__ __launch_bounds__(256) void gcn_init_kernel(const float* __restrict__ m,
                                                       const float* __restrict__ norm,
                                                       const float* __restrict__ bgl,
                                                       float* __restrict__ hN) {
  int idx = blockIdx.x * 256 + threadIdx.x;
  if (idx >= TN * DCH) return;
  int t = idx >> 7, d = idx & 127;
  float nv = norm[t];
  hN[idx] = m[idx] * nv * nv + bgl[d];
}

__global__ __launch_bounds__(128) void gcn_scatter_kernel(const int* __restrict__ edges,
                                                          const float* __restrict__ m,
                                                          const float* __restrict__ norm,
                                                          float* __restrict__ hN) {
  int e = blockIdx.x;
  int b = e / EE, le = e - b * EE;
  int src = edges[b * 2 * EE + le] + b * NN;
  int dst = edges[b * 2 * EE + EE + le] + b * NN;
  float coef = norm[src] * norm[dst];
  int d = threadIdx.x;
  atomicAdd(&hN[(size_t)dst * DCH + d], m[(size_t)src * DCH + d] * coef);
}

__global__ __launch_bounds__(256) void tanh_kernel(const float* __restrict__ src,
                                                   float* __restrict__ dst, int ldc) {
  int idx = blockIdx.x * 256 + threadIdx.x;
  if (idx >= TN * DCH) return;
  int t = idx >> 7, d = idx & 127;
  dst[(size_t)t * ldc + d] = tanhf(src[idx]);
}

// =====================================================================
// WMMA GEMM: C[M,128] = A(f32,[M,128]) @ BT(bf16,[n][k]) + bias
// =====================================================================
__global__ __launch_bounds__(256)
void gemm_nodes_kernel(const float* __restrict__ A, const bf16_t* __restrict__ BT,
                       const float* __restrict__ bias,
                       float* __restrict__ outF, bf16_t* __restrict__ outB, int ldc) {
  const int lane = threadIdx.x & 31;
  const int wav  = threadIdx.x >> 5;
  const int Mbase = blockIdx.x * 128 + wav * 16;
  const int mrow = lane & 15;
  const int hi   = lane >> 4;
  const int sa = hi ? 8 : 0;
  const int sb = hi ? 16 : 0;

  const float* arow = A + (size_t)(Mbase + mrow) * DCH;
  v8f acc[8];
#pragma unroll
  for (int nt = 0; nt < 8; ++nt) acc[nt] = vzero8();

#pragma unroll
  for (int kk = 0; kk < 4; ++kk) {
    const float* ap = arow + kk * 32;
    v4f a0 = *(const v4f*)(ap + sa);
    v4f a1 = *(const v4f*)(ap + sa + 4);
    v4f a2 = *(const v4f*)(ap + sa + 16);
    v4f a3 = *(const v4f*)(ap + sa + 20);
    v16bf av;
#pragma unroll
    for (int i = 0; i < 4; ++i) {
      av[i]      = (bf16_t)a0[i];
      av[4 + i]  = (bf16_t)a1[i];
      av[8 + i]  = (bf16_t)a2[i];
      av[12 + i] = (bf16_t)a3[i];
    }
#pragma unroll
    for (int nt = 0; nt < 8; ++nt) {
      const bf16_t* bp = BT + (size_t)(nt * 16 + mrow) * DCH + kk * 32 + sa;
      v16bf bv = combine16(*(const v8bf*)bp, *(const v8bf*)(bp + 16));
      acc[nt] = __builtin_amdgcn_wmma_f32_16x16x32_bf16(
          false, av, false, bv, (short)0, acc[nt], false, false);
    }
  }

#pragma unroll
  for (int nt = 0; nt < 8; ++nt) {
    int n = nt * 16 + mrow;
    float bn = bias ? bias[n] : 0.f;
#pragma unroll
    for (int r = 0; r < 8; ++r) {
      int row = Mbase + r + hi * 8;
      float val = acc[nt][r] + bn;
      if (outF) outF[(size_t)row * ldc + n] = val;
      else      outB[(size_t)row * DCH + n] = (bf16_t)val;
    }
  }
}

// =====================================================================
// Flash attention, LDS-staged K/V shared by 8 waves per block.
// Q,K bf16 [TN,128]; VT bf16 [128][TN]; out f32 [TN,128] (= softmax(QK^T)@V)
// 256 threads = 8 waves; block covers 128 rows; j streamed in 32-col tiles.
// =====================================================================
__global__ __launch_bounds__(256)
void flash_attn_kernel(const bf16_t* __restrict__ Q, const bf16_t* __restrict__ K,
                       const bf16_t* __restrict__ VT, float* __restrict__ Aout) {
#if defined(HAS_ASYNC_LDS)
#define NKBUF 2
#else
#define NKBUF 1
#endif
  __shared__ __align__(32) bf16_t kbuf[NKBUF][32][136];   // 272-B padded rows
  __shared__ __align__(32) bf16_t vbuf[NKBUF][128][40];   // 80-B padded rows
  __shared__ __align__(32) bf16_t pbuf[8][16][40];        // per-wave P staging

  const int tid  = threadIdx.x;
  const int lane = tid & 31;
  const int wav  = tid >> 5;
  const int Mbase = blockIdx.x * 128 + wav * 16;
  const int mrow = lane & 15;
  const int hi   = lane >> 4;
  const int sa  = hi ? 8 : 0;    // acc/P row offset of this lane half
  const int sbE = hi ? 16 : 0;   // B-side K offset (elements)

  // cooperative-copy chunk coordinates for this thread (2 chunks each for K and V)
  const int kr0 = tid >> 4,          ks0 = tid & 15;          // K chunk 0
  const int kr1 = (tid + 256) >> 4,  ks1 = tid & 15;          // K chunk 1
  const int vn0 = tid >> 2,          vs0 = tid & 3;           // V chunk 0
  const int vn1 = (tid + 256) >> 2,  vs1 = tid & 3;           // V chunk 1

#if defined(HAS_ASYNC_LDS)
  auto issue_tile = [&](int buf, int j0) {
    const bf16_t* gk0 = K + (size_t)(j0 + kr0) * DCH + ks0 * 8;
    const bf16_t* gk1 = K + (size_t)(j0 + kr1) * DCH + ks1 * 8;
    __builtin_amdgcn_global_load_async_to_lds_b128(
        (gas_p)gk0, (las_p)&kbuf[buf][kr0][ks0 * 8], 0, 0);
    __builtin_amdgcn_global_load_async_to_lds_b128(
        (gas_p)gk1, (las_p)&kbuf[buf][kr1][ks1 * 8], 0, 0);
    const bf16_t* gv0 = VT + (size_t)vn0 * TN + j0 + vs0 * 8;
    const bf16_t* gv1 = VT + (size_t)vn1 * TN + j0 + vs1 * 8;
    __builtin_amdgcn_global_load_async_to_lds_b128(
        (gas_p)gv0, (las_p)&vbuf[buf][vn0][vs0 * 8], 0, 0);
    __builtin_amdgcn_global_load_async_to_lds_b128(
        (gas_p)gv1, (las_p)&vbuf[buf][vn1][vs1 * 8], 0, 0);
  };
#endif

  // Q A-operands (held in registers for the whole row-tile)
  v16bf qa[4];
  const bf16_t* qrow = Q + (size_t)(Mbase + mrow) * DCH;
#pragma unroll
  for (int kk = 0; kk < 4; ++kk)
    qa[kk] = combine16(*(const v8bf*)(qrow + kk * 32 + sbE),
                       *(const v8bf*)(qrow + kk * 32 + sbE + 8));

  v8f acc[8];
#pragma unroll
  for (int nt = 0; nt < 8; ++nt) acc[nt] = vzero8();
  float m_s[8], l_s[8];
#pragma unroll
  for (int r = 0; r < 8; ++r) { m_s[r] = -INFINITY; l_s[r] = 0.f; }

  bf16_t (*pb)[40] = pbuf[wav];
  const int NTI = TN / 32;

#if defined(HAS_ASYNC_LDS)
  issue_tile(0, 0);
#endif

  for (int jt = 0; jt < NTI; ++jt) {
    const int j0 = jt * 32;
#if defined(HAS_ASYNC_LDS)
    const int cur = jt & 1;
    if (jt + 1 < NTI) {
      issue_tile(1 - cur, j0 + 32);
      ASYNC_WAIT(4);          // tile jt fully landed (only next tile in flight)
    } else {
      ASYNC_WAIT(0);
    }
    __syncthreads();          // all waves' portions visible
#else
    const int cur = 0;
    *(v4u*)&kbuf[0][kr0][ks0 * 8] = *(const v4u*)(K + (size_t)(j0 + kr0) * DCH + ks0 * 8);
    *(v4u*)&kbuf[0][kr1][ks1 * 8] = *(const v4u*)(K + (size_t)(j0 + kr1) * DCH + ks1 * 8);
    *(v4u*)&vbuf[0][vn0][vs0 * 8] = *(const v4u*)(VT + (size_t)vn0 * TN + j0 + vs0 * 8);
    *(v4u*)&vbuf[0][vn1][vs1 * 8] = *(const v4u*)(VT + (size_t)vn1 * TN + j0 + vs1 * 8);
    __syncthreads();
#endif

    // ---- S = Q @ K^T for 32 columns (B operands from LDS) ----
    v8f s0 = vzero8(), s1 = vzero8();
#pragma unroll
    for (int kk = 0; kk < 4; ++kk) {
      const bf16_t* kp0 = &kbuf[cur][mrow][kk * 32 + sbE];
      v16bf b0 = combine16(*(const v8bf*)kp0, *(const v8bf*)(kp0 + 8));
      s0 = __builtin_amdgcn_wmma_f32_16x16x32_bf16(false, qa[kk], false, b0,
                                                   (short)0, s0, false, false);
      const bf16_t* kp1 = &kbuf[cur][16 + mrow][kk * 32 + sbE];
      v16bf b1 = combine16(*(const v8bf*)kp1, *(const v8bf*)(kp1 + 8));
      s1 = __builtin_amdgcn_wmma_f32_16x16x32_bf16(false, qa[kk], false, b1,
                                                   (short)0, s1, false, false);
    }

    // ---- online softmax update ----
    float alpha[8];
    int need = 0;
#pragma unroll
    for (int r = 0; r < 8; ++r) {
      float t = fmaxf(s0[r], s1[r]);
      t = fmaxf(t, __shfl_xor(t, 1));
      t = fmaxf(t, __shfl_xor(t, 2));
      t = fmaxf(t, __shfl_xor(t, 4));
      t = fmaxf(t, __shfl_xor(t, 8));
      float mo = m_s[r];
      float mn = fmaxf(mo, t);
      alpha[r] = __expf(mo - mn);
      need |= (mn > mo) ? 1 : 0;
      m_s[r] = mn;
      float p0 = __expf(s0[r] - mn);
      float p1 = __expf(s1[r] - mn);
      float rs = p0 + p1;
      rs += __shfl_xor(rs, 1);
      rs += __shfl_xor(rs, 2);
      rs += __shfl_xor(rs, 4);
      rs += __shfl_xor(rs, 8);
      l_s[r] = l_s[r] * alpha[r] + rs;
      int prow = r + sa;
      pb[prow][mrow]      = (bf16_t)p0;
      pb[prow][16 + mrow] = (bf16_t)p1;
    }
    // rescale accumulators only when some row max moved (wave-uniform branch)
    if (__any(need)) {
#pragma unroll
      for (int r = 0; r < 8; ++r) {
        float a = alpha[r];
#pragma unroll
        for (int nt = 0; nt < 8; ++nt) acc[nt][r] *= a;
      }
    }

    // ---- reload P in A-operand layout (per-wave LDS region, DS in-order) ----
    v16bf pa = combine16(*(const v8bf*)(&pb[mrow][sbE]),
                         *(const v8bf*)(&pb[mrow][sbE + 8]));

    // ---- acc += P @ V (K = 32 streamed columns; V from LDS) ----
#pragma unroll
    for (int nt = 0; nt < 8; ++nt) {
      const bf16_t* vp = &vbuf[cur][nt * 16 + mrow][sbE];
      v16bf bv = combine16(*(const v8bf*)vp, *(const v8bf*)(vp + 8));
      acc[nt] = __builtin_amdgcn_wmma_f32_16x16x32_bf16(false, pa, false, bv,
                                                        (short)0, acc[nt], false, false);
    }
    __syncthreads();          // guard K/V buffer reuse
  }

  // ---- finalize ----
#pragma unroll
  for (int nt = 0; nt < 8; ++nt) {
    int n = nt * 16 + mrow;
#pragma unroll
    for (int r = 0; r < 8; ++r) {
      int row = Mbase + r + hi * 8;
      Aout[(size_t)row * DCH + n] = acc[nt][r] / l_s[r];
    }
  }
}

// =====================================================================
// host-side launch
// =====================================================================
extern "C" void kernel_launch(void* const* d_in, const int* in_sizes, int n_in,
                              void* d_out, int out_size, void* d_ws, size_t ws_size,
                              hipStream_t stream) {
  const float* x     = (const float*)d_in[0];
  const int*   edges = (const int*)d_in[1];
  const float* Wg    = (const float*)d_in[2];
  const float* bg    = (const float*)d_in[3];
  const float* Wq    = (const float*)d_in[4];
  const float* bq    = (const float*)d_in[5];
  const float* Wk    = (const float*)d_in[6];
  const float* bk    = (const float*)d_in[7];
  const float* Wv    = (const float*)d_in[8];
  const float* bv    = (const float*)d_in[9];
  const float* Wo    = (const float*)d_in[10];
  const float* bo    = (const float*)d_in[11];
  const float* Wc    = (const float*)d_in[12];
  const float* bc    = (const float*)d_in[13];
  float* out = (float*)d_out;
  char*  ws  = (char*)d_ws;

  const size_t MB = 1u << 20;
  float*  hA   = (float*)(ws + 0 * MB);
  float*  mbuf = (float*)(ws + 4 * MB);
  float*  hN   = (float*)(ws + 8 * MB);
  float*  att  = (float*)(ws + 12 * MB);         // also v-f32 temp
  bf16_t* qbf  = (bf16_t*)(ws + 16 * MB);
  bf16_t* kbf  = (bf16_t*)(ws + 18 * MB);
  bf16_t* vT   = (bf16_t*)(ws + 20 * MB);
  float*  norm = (float*)(ws + 22 * MB);
  float*  Wsum = (float*)(ws + 22 * MB + 32 * 1024);
  float*  W2   = (float*)(ws + 22 * MB + 96 * 1024);
  float*  b2   = (float*)(ws + 22 * MB + 160 * 1024);
  bf16_t* WqT  = (bf16_t*)(ws + 22 * MB + 192 * 1024);
  bf16_t* WkT  = (bf16_t*)(ws + 22 * MB + 224 * 1024);
  bf16_t* WvT  = (bf16_t*)(ws + 22 * MB + 256 * 1024);
  bf16_t* W2T  = (bf16_t*)(ws + 22 * MB + 288 * 1024);
  bf16_t* WgT  = (bf16_t*)(ws + 22 * MB + 320 * 1024);

  // ---- weight prep ----
  wsum_kernel<<<64, 256, 0, stream>>>(Wc, Wsum);
  w2_kernel<<<128, 128, 0, stream>>>(Wo, Wsum, W2);
  b2_kernel<<<1, 128, 0, stream>>>(bo, Wsum, bc, b2);
  wtrans_kernel<<<128, 128, 0, stream>>>(Wq, WqT);
  wtrans_kernel<<<128, 128, 0, stream>>>(Wk, WkT);
  wtrans_kernel<<<128, 128, 0, stream>>>(Wv, WvT);
  wtrans_kernel<<<128, 128, 0, stream>>>(W2, W2T);
  for (int l = 0; l < LL; ++l)
    wtrans_kernel<<<128, 128, 0, stream>>>(Wg + (size_t)l * DCH * DCH,
                                           WgT + (size_t)l * DCH * DCH);

  // ---- GCN normalization coefficients ----
  deg_init_kernel<<<32, 256, 0, stream>>>(norm);
  deg_scatter_kernel<<<(BG * EE + 255) / 256, 256, 0, stream>>>(edges, norm);
  rsqrt_kernel<<<32, 256, 0, stream>>>(norm);

  // ---- attention path ----
  gemm_nodes_kernel<<<64, 256, 0, stream>>>(x, WqT, bq, nullptr, qbf, DCH);
  gemm_nodes_kernel<<<64, 256, 0, stream>>>(x, WkT, bk, nullptr, kbf, DCH);
  gemm_nodes_kernel<<<64, 256, 0, stream>>>(x, WvT, bv, att, nullptr, DCH);
  vtrans_kernel<<<TN, 128, 0, stream>>>(att, vT);
  flash_attn_kernel<<<TN / 128, 256, 0, stream>>>(qbf, kbf, vT, att);
  gemm_nodes_kernel<<<64, 256, 0, stream>>>(att, W2T, b2, out + DCH, nullptr, 2 * DCH);

  // ---- GCN path ----
  const float* hin = x;
  for (int l = 0; l < LL; ++l) {
    gemm_nodes_kernel<<<64, 256, 0, stream>>>(hin, WgT + (size_t)l * DCH * DCH,
                                              nullptr, mbuf, nullptr, DCH);
    gcn_init_kernel<<<(TN * DCH) / 256, 256, 0, stream>>>(mbuf, norm, bg + l * DCH, hN);
    gcn_scatter_kernel<<<BG * EE, 128, 0, stream>>>(edges, mbuf, norm, hN);
    if (l < LL - 1) {
      tanh_kernel<<<(TN * DCH) / 256, 256, 0, stream>>>(hN, hA, DCH);
      hin = hA;
    } else {
      tanh_kernel<<<(TN * DCH) / 256, 256, 0, stream>>>(hN, out, 2 * DCH);
    }
  }
}
